// DynamicRouting_75617194213513
// MI455X (gfx1250) — matrix operand order repaired and checked by reading, not compiled, not run on hardware
//
#include <hip/hip_runtime.h>

// CDNA5 / gfx1250 fused grouped-1x1-conv + dynamic routing.
// Phase 1: per-group GEMM con = W_g (128x256) x X_g (256 x 32 pixels) via
//          v_wmma_f32_16x16x32_bf16 (fp32 inputs converted to bf16 in-reg).
// Phase 2: routing iterations entirely in LDS (con tile stays on-chip).

typedef __attribute__((ext_vector_type(16))) __bf16 v16bf;
typedef __attribute__((ext_vector_type(8)))  float  v8f;

constexpr int kG   = 8;     // groups
constexpr int kFI  = 256;   // in channels per group
constexpr int kFO  = 128;   // out channels per group
constexpr int kHW  = 4096;  // H*W
constexpr int kB   = 8;     // batch
constexpr int kNT  = 32;    // pixels per workgroup
constexpr int kCS  = 34;    // con LDS row stride (padded: 8*34 % 64 == 16 -> no bank clash)

__global__ __launch_bounds__(256)
void dynrout_fused_kernel(const float* __restrict__ x,
                          const float* __restrict__ w,
                          const float* __restrict__ bias,
                          float* __restrict__ out)
{
    extern __shared__ float smem[];
    float* conS  = smem;                          // [G][FO][kCS]
    float* vS    = conS + kG * kFO * kCS;         // [FO][kNT]
    float* betaS = vS   + kFO * kNT;              // [G][kNT]

    const int tid  = threadIdx.x;
    const int wave = tid >> 5;        // one wave == one group g
    const int lane = tid & 31;
    const int half = lane >> 4;       // lane half (ISA fragment layouts)
    const int ln   = lane & 15;

    const long long pix0 = (long long)blockIdx.x * kNT;
    const int b   = (int)(pix0 >> 12);        // / 4096
    const int hw0 = (int)(pix0 & (kHW - 1));  // contiguous 32-pixel tile in one image

    // ---------------- Phase 1: GEMM (bf16 WMMA) ----------------
    v8f acc[8][2] = {};   // 8 M-tiles (FO=128) x 2 N-tiles (32 pixels)

    const float* xg = x + ((size_t)(b * (kG * kFI) + wave * kFI)) * kHW + hw0;
    const float* wg = w + (size_t)wave * kFO * kFI;

    for (int kc = 0; kc < kFI / 32; ++kc) {
        // B fragments: X tile [K=32 x N=16] per subtile.
        // Layout: element e -> K = (e&7) + half*8 + (e>>3)*16 (col n=lane%16), matching A.
        v16bf B0, B1;
        const float* xb = xg + (size_t)kc * 32 * kHW;
#pragma unroll
        for (int e = 0; e < 16; ++e) {
            const float* xr = xb + (size_t)((e & 7) + half * 8 + (e >> 3) * 16) * kHW;
            B0[e] = (__bf16)xr[ln];         // pixels 0..15
            B1[e] = (__bf16)xr[ln + 16];    // pixels 16..31
        }
#pragma unroll
        for (int m = 0; m < 8; ++m) {
            // A fragment: W tile [M=16 x K=32]; lane row = m*16+ln,
            // element e -> K = (e&7) + half*8 + (e>>3)*16  (ISA 16-bit A layout)
            const float* wr = wg + (size_t)(m * 16 + ln) * kFI + kc * 32 + half * 8;
            float4 w0 = *(const float4*)(wr +  0);
            float4 w1 = *(const float4*)(wr +  4);
            float4 w2 = *(const float4*)(wr + 16);
            float4 w3 = *(const float4*)(wr + 20);
            v16bf A;
            A[ 0]=(__bf16)w0.x; A[ 1]=(__bf16)w0.y; A[ 2]=(__bf16)w0.z; A[ 3]=(__bf16)w0.w;
            A[ 4]=(__bf16)w1.x; A[ 5]=(__bf16)w1.y; A[ 6]=(__bf16)w1.z; A[ 7]=(__bf16)w1.w;
            A[ 8]=(__bf16)w2.x; A[ 9]=(__bf16)w2.y; A[10]=(__bf16)w2.z; A[11]=(__bf16)w2.w;
            A[12]=(__bf16)w3.x; A[13]=(__bf16)w3.y; A[14]=(__bf16)w3.z; A[15]=(__bf16)w3.w;

            acc[m][0] = __builtin_amdgcn_wmma_f32_16x16x32_bf16(
                false, A, false, B0, (short)0, acc[m][0], false, false);
            acc[m][1] = __builtin_amdgcn_wmma_f32_16x16x32_bf16(
                false, A, false, B1, (short)0, acc[m][1], false, false);
        }
    }

    // Spill con tile to LDS. C/D layout: VGPR r holds M = r + half*8.
#pragma unroll
    for (int m = 0; m < 8; ++m)
#pragma unroll
        for (int nt = 0; nt < 2; ++nt)
#pragma unroll
            for (int r = 0; r < 8; ++r) {
                int f = m * 16 + half * 8 + r;
                int n = nt * 16 + ln;
                conS[(wave * kFO + f) * kCS + n] = acc[m][nt][r];
            }
    betaS[tid] = 0.0f;   // 8*32 == 256 entries, one per thread
    __syncthreads();

    // ---------------- Phase 2: routing (in LDS) ----------------
    const int n  = tid & (kNT - 1);   // pixel within tile (lane-contiguous)
    const int fb = tid >> 5;          // f block (== wave id) -> owns f = fb*16..fb*16+15

    for (int it = 0; it < 3; ++it) {
        float al[kG];
#pragma unroll
        for (int g2 = 0; g2 < kG; ++g2) {
            float bta = betaS[g2 * kNT + n];
            al[g2] = 1.0f / (1.0f + __expf(-bta));
        }
        float vloc[16];
#pragma unroll
        for (int j = 0; j < 16; ++j) {
            int f = fb * 16 + j;
            float s = 0.0f;
#pragma unroll
            for (int g2 = 0; g2 < kG; ++g2)
                s += al[g2] * conS[(g2 * kFO + f) * kCS + n];
            vloc[j] = s;
        }
        if (it == 2) {
            // final: out[b][f][hw] = v + bias[f]; 128B-contiguous per wave-store
            size_t obase = (size_t)b * kFO * kHW + (size_t)hw0 + n;
#pragma unroll
            for (int j = 0; j < 16; ++j) {
                int f = fb * 16 + j;
                out[obase + (size_t)f * kHW] = vloc[j] + bias[f];
            }
        } else {
#pragma unroll
            for (int j = 0; j < 16; ++j)
                vS[(fb * 16 + j) * kNT + n] = vloc[j];
            __syncthreads();
            // beta[g][n] += sum_f v[f][n] * con[g][f][n]; thread owns (g=fb, n)
            float d = 0.0f;
            for (int f = 0; f < kFO; ++f)
                d += vS[f * kNT + n] * conS[(fb * kFO + f) * kCS + n];
            betaS[fb * kNT + n] += d;
            __syncthreads();
        }
    }
}

extern "C" void kernel_launch(void* const* d_in, const int* in_sizes, int n_in,
                              void* d_out, int out_size, void* d_ws, size_t ws_size,
                              hipStream_t stream) {
    const float* x    = (const float*)d_in[0];   // [8, 2048, 64, 64] f32
    const float* w    = (const float*)d_in[1];   // [1024, 256, 1, 1] f32
    const float* bias = (const float*)d_in[2];   // [128] f32
    float* out        = (float*)d_out;           // [8, 128, 64, 64] f32

    (void)in_sizes; (void)n_in; (void)out_size; (void)d_ws; (void)ws_size;

    const int nblocks = (kB * kHW) / kNT;  // 32768 / 32 = 1024 workgroups
    const size_t smem = (size_t)(kG * kFO * kCS + kFO * kNT + kG * kNT) * sizeof(float);

    dynrout_fused_kernel<<<nblocks, 256, smem, stream>>>(x, w, bias, out);
}
